// Attention_13262859010647
// MI455X (gfx1250) — matrix operand order repaired and checked by reading, not compiled, not run on hardware
//
#include <hip/hip_runtime.h>
#include <hip/hip_bf16.h>
#include <math.h>
#include <stdint.h>

typedef __attribute__((ext_vector_type(16))) __bf16 v16bf;
typedef __attribute__((ext_vector_type(2)))  __bf16 v2bf;
typedef __attribute__((ext_vector_type(8)))  float  v8f;
typedef __attribute__((ext_vector_type(4)))  int    v4i;

#define D_MODEL 768
#define SEQ     4096
#define D_HEAD  64

#define AS1 __attribute__((address_space(1)))
#define AS3 __attribute__((address_space(3)))

#if defined(__AMDGCN__) && __has_builtin(__builtin_amdgcn_global_load_async_to_lds_b128) && \
    __has_builtin(__builtin_amdgcn_s_wait_asynccnt)
#define USE_ASYNC_LDS 1
#else
#define USE_ASYNC_LDS 0
#endif

union Frag16 {
    v16bf v;
    unsigned short u[16];
    unsigned int   w[8];
    uint4 q[2];
};

// Scalar f32 -> bf16, round-half-up (2 VALU).
__device__ __forceinline__ unsigned short f2bf(float f) {
    union { float f; unsigned int u; } x;
    x.f = f;
    return (unsigned short)((x.u + 0x8000u) >> 16);
}

// Pack two floats into one dword of bf16.  v_cvt_pk_bf16_f32 when the
// toolchain has it; otherwise round-half-up + single v_perm_b32 pack.
__device__ __forceinline__ unsigned int f2bf2(float lo, float hi) {
#if __has_builtin(__builtin_amdgcn_cvt_pk_bf16_f32)
    union { v2bf v; unsigned int u; } x;
    x.v = __builtin_amdgcn_cvt_pk_bf16_f32(lo, hi);
    return x.u;
#else
    union { float f; unsigned int u; } a, b;
    a.f = lo; b.f = hi;
    // result[15:0] = (a+0x8000)[31:16], result[31:16] = (b+0x8000)[31:16]
    return __builtin_amdgcn_perm(b.u + 0x8000u, a.u + 0x8000u, 0x07060302u);
#endif
}

// -------------------------------------------------------------------------
// Kernel 0: one-time f32 -> bf16 conversion (8 elements / thread).
// Amortizes conversion cost once instead of per GEMM tile.
// -------------------------------------------------------------------------
__global__ __launch_bounds__(256) void cvt_bf16_kernel(
    const float* __restrict__ src, unsigned short* __restrict__ dst, int n8)
{
    const int i = blockIdx.x * blockDim.x + threadIdx.x;
    if (i >= n8) return;
    const float4* s = (const float4*)src + i * 2;
    const float4 x0 = s[0];
    const float4 x1 = s[1];
    uint4 o;
    o.x = f2bf2(x0.x, x0.y);
    o.y = f2bf2(x0.z, x0.w);
    o.z = f2bf2(x1.x, x1.y);
    o.w = f2bf2(x1.z, x1.w);
    ((uint4*)dst)[i] = o;
}

// -------------------------------------------------------------------------
// Kernel 1: fused Q/K/V projection.  Y = X @ W^T + b  -> bf16, head-major
// [head][seq][64].  One wave computes a 32x32 tile (2x2 register tiling:
// 2 A-frags x 2 B-frags -> 4 wmma per K-step of 32), halving operand
// loads per wmma vs a 16x16 tile.
// -------------------------------------------------------------------------
__global__ __launch_bounds__(256) void qkv_proj_kernel(
    const unsigned short* __restrict__ Xq, const unsigned short* __restrict__ Xk,
    const unsigned short* __restrict__ Xv,
    const unsigned short* __restrict__ Wqb, const unsigned short* __restrict__ Wkb,
    const unsigned short* __restrict__ Wvb,
    const float* __restrict__ bq, const float* __restrict__ bk,
    const float* __restrict__ bv,
    unsigned short* __restrict__ Qh, unsigned short* __restrict__ Kh,
    unsigned short* __restrict__ Vh)
{
    const int lane = threadIdx.x & 31;
    const int wave = threadIdx.x >> 5;
    const int row  = lane & 15;
    const int hf   = lane >> 4;

    const unsigned short* X; const unsigned short* W; const float* Bv;
    unsigned short* Out;
    if (blockIdx.z == 0)      { X = Xq; W = Wqb; Bv = bq; Out = Qh; }
    else if (blockIdx.z == 1) { X = Xk; W = Wkb; Bv = bk; Out = Kh; }
    else                      { X = Xv; W = Wvb; Bv = bv; Out = Vh; }

    const int mb = blockIdx.x * 32;
    const int nb = (blockIdx.y * 8 + wave) * 32;

    const unsigned short* arow0 = X + (mb + row) * D_MODEL;       // A rows 0..15
    const unsigned short* arow1 = X + (mb + 16 + row) * D_MODEL;  // A rows 16..31
    const unsigned short* brow0 = W + (nb + row) * D_MODEL;       // B cols 0..15
    const unsigned short* brow1 = W + (nb + 16 + row) * D_MODEL;  // B cols 16..31

    const int c0 = hf * 8;    // A k-offset within fragment (per 16-lane half)
    const int kb = hf * 8;    // B k-offset within fragment

    v8f acc00 = {}, acc01 = {}, acc10 = {}, acc11 = {};
    for (int k0 = 0; k0 < D_MODEL; k0 += 32) {
        Frag16 a0, a1, b0, b1;
        a0.q[0] = *(const uint4*)(arow0 + k0 + c0);
        a0.q[1] = *(const uint4*)(arow0 + k0 + c0 + 16);
        a1.q[0] = *(const uint4*)(arow1 + k0 + c0);
        a1.q[1] = *(const uint4*)(arow1 + k0 + c0 + 16);
        b0.q[0] = *(const uint4*)(brow0 + k0 + kb);
        b0.q[1] = *(const uint4*)(brow0 + k0 + kb + 16);
        b1.q[0] = *(const uint4*)(brow1 + k0 + kb);
        b1.q[1] = *(const uint4*)(brow1 + k0 + kb + 16);
        acc00 = __builtin_amdgcn_wmma_f32_16x16x32_bf16(
            false, a0.v, false, b0.v, (short)0, acc00, false, false);
        acc01 = __builtin_amdgcn_wmma_f32_16x16x32_bf16(
            false, a0.v, false, b1.v, (short)0, acc01, false, false);
        acc10 = __builtin_amdgcn_wmma_f32_16x16x32_bf16(
            false, a1.v, false, b0.v, (short)0, acc10, false, false);
        acc11 = __builtin_amdgcn_wmma_f32_16x16x32_bf16(
            false, a1.v, false, b1.v, (short)0, acc11, false, false);
    }

    const int n0 = nb + row;
    const int n1 = nb + 16 + row;
    const float bb0 = Bv[n0];
    const float bb1 = Bv[n1];
    unsigned short* ob0 = Out + (n0 >> 6) * (SEQ * D_HEAD) + (n0 & 63);
    unsigned short* ob1 = Out + (n1 >> 6) * (SEQ * D_HEAD) + (n1 & 63);
    #pragma unroll
    for (int r = 0; r < 8; ++r) {
        const int m0 = mb + r + hf * 8;       // C layout: lanes>=16 hold M+8
        const int m1 = m0 + 16;
        ob0[m0 * D_HEAD] = f2bf(acc00[r] + bb0);
        ob1[m0 * D_HEAD] = f2bf(acc01[r] + bb1);
        ob0[m1 * D_HEAD] = f2bf(acc10[r] + bb0);
        ob1[m1 * D_HEAD] = f2bf(acc11[r] + bb1);
    }
}

// -------------------------------------------------------------------------
// Kernel 2: causal flash attention per head, bf16 WMMA, fp32 accumulators.
// Block = 4 waves x 16 q-rows (64 rows); kv tiles of 32 staged in LDS.
// K tile staged with GLOBAL_LOAD_ASYNC_TO_LDS_B128 (ASYNCcnt).
// -------------------------------------------------------------------------
__global__ __launch_bounds__(128) void attn_kernel(
    const unsigned short* __restrict__ Qh,
    const unsigned short* __restrict__ Kh,
    const unsigned short* __restrict__ Vh,
    unsigned short* __restrict__ Obuf)
{
    __shared__ uint4          kbuf[256];        // K tile: 32 x 64 bf16, row-major
    __shared__ unsigned short vtbuf[64 * 32];   // V tile transposed: [d][kv]
    __shared__ unsigned short pbuf[4][16 * 32]; // per-wave P scratch (C->A relayout)

    const int lane  = threadIdx.x & 31;
    const int wave  = threadIdx.x >> 5;
    const int row   = lane & 15;
    const int hf    = lane >> 4;
    const int h     = blockIdx.y;
    const int qb    = blockIdx.x;
    const int qbase = qb * 64 + wave * 16;

    const unsigned short* Qp = Qh + h * (SEQ * D_HEAD);
    const unsigned short* Kp = Kh + h * (SEQ * D_HEAD);
    const unsigned short* Vp = Vh + h * (SEQ * D_HEAD);

    // Q A-fragments, d = 0..31 and 32..63 (stay in registers all loop long)
    Frag16 qa0, qa1;
    {
        const unsigned short* qr = Qp + (qbase + row) * D_HEAD;
        const int c0 = hf * 8;
        qa0.q[0] = *(const uint4*)(qr + c0);
        qa0.q[1] = *(const uint4*)(qr + c0 + 16);
        qa1.q[0] = *(const uint4*)(qr + 32 + c0);
        qa1.q[1] = *(const uint4*)(qr + 32 + c0 + 16);
    }

    v8f o0 = {}, o1 = {}, o2 = {}, o3 = {};
    float runmax[8], runsum[8];
    #pragma unroll
    for (int r = 0; r < 8; ++r) { runmax[r] = -1e30f; runsum[r] = 0.0f; }

    const int jtiles = qb * 2 + 2;               // causal: kv0 <= qb*64+63
    for (int j = 0; j < jtiles; ++j) {
        const int kv0 = j * 32;

        // ---- cooperative stage: K row-major (async DMA), V transposed ----
        {
            const uint4* ksrc = (const uint4*)(Kp + kv0 * D_HEAD);
#if USE_ASYNC_LDS
            // gfx1250 async global->LDS copy, tracked by ASYNCcnt.
            __builtin_amdgcn_global_load_async_to_lds_b128(
                (AS1 v4i*)(uintptr_t)(ksrc + threadIdx.x * 2 + 0),
                (AS3 v4i*)(void*)(kbuf + threadIdx.x * 2 + 0), 0, 0);
            __builtin_amdgcn_global_load_async_to_lds_b128(
                (AS1 v4i*)(uintptr_t)(ksrc + threadIdx.x * 2 + 1),
                (AS3 v4i*)(void*)(kbuf + threadIdx.x * 2 + 1), 0, 0);
#else
            kbuf[threadIdx.x * 2 + 0] = ksrc[threadIdx.x * 2 + 0];
            kbuf[threadIdx.x * 2 + 1] = ksrc[threadIdx.x * 2 + 1];
#endif

            const uint4* vsrc = (const uint4*)(Vp + kv0 * D_HEAD);
            union { uint4 q; unsigned short u[8]; } p0, p1;
            p0.q = vsrc[threadIdx.x * 2 + 0];
            p1.q = vsrc[threadIdx.x * 2 + 1];
            const int e  = threadIdx.x * 16;     // flat bf16 element index
            const int kr = e >> 6;               // kv row within tile
            const int db = e & 63;               // d base
            #pragma unroll
            for (int i = 0; i < 8; ++i) vtbuf[(db + i) * 32 + kr]     = p0.u[i];
            #pragma unroll
            for (int i = 0; i < 8; ++i) vtbuf[(db + 8 + i) * 32 + kr] = p1.u[i];

            if (j + 1 < jtiles) {                 // gfx1250 global_prefetch path
                __builtin_prefetch((const void*)(Kp + (kv0 + 32) * D_HEAD), 0, 1);
                __builtin_prefetch((const void*)(Vp + (kv0 + 32) * D_HEAD), 0, 1);
            }
        }
#if USE_ASYNC_LDS
        __builtin_amdgcn_s_wait_asynccnt(0);
#endif
        __syncthreads();

        // ---- scores: S = Q K^T  (two 16x16 C tiles, K-dim 64 = 2 x wmma) ----
        v8f s0v = {}, s1v = {};
        #pragma unroll
        for (int ns = 0; ns < 2; ++ns) {
            const int nloc = ns * 16 + row;       // lane's kv column
            Frag16 b0, b1;
            b0.q[0] = kbuf[nloc * 8 + hf + 0];       // d = 0..31
            b0.q[1] = kbuf[nloc * 8 + hf + 2];
            b1.q[0] = kbuf[nloc * 8 + 4 + hf + 0];   // d = 32..63
            b1.q[1] = kbuf[nloc * 8 + 4 + hf + 2];
            v8f s = {};
            s = __builtin_amdgcn_wmma_f32_16x16x32_bf16(
                false, qa0.v, false, b0.v, (short)0, s, false, false);
            s = __builtin_amdgcn_wmma_f32_16x16x32_bf16(
                false, qa1.v, false, b1.v, (short)0, s, false, false);
            if (ns == 0) s0v = s; else s1v = s;
        }

        // ---- online softmax (row reductions across 16-lane halves) ----
        #pragma unroll
        for (int r = 0; r < 8; ++r) {
            const int qrow = qbase + r + hf * 8;
            float x0 = s0v[r] * 0.125f;           // 1/sqrt(64)
            float x1 = s1v[r] * 0.125f;
            if (kv0 + row      > qrow) x0 = -1e30f;
            if (kv0 + 16 + row > qrow) x1 = -1e30f;
            float m = fmaxf(x0, x1);
            m = fmaxf(m, __shfl_xor(m, 1));
            m = fmaxf(m, __shfl_xor(m, 2));
            m = fmaxf(m, __shfl_xor(m, 4));
            m = fmaxf(m, __shfl_xor(m, 8));
            const float nmax  = fmaxf(runmax[r], m);
            const float scale = __expf(runmax[r] - nmax);
            runmax[r] = nmax;
            const float p0 = __expf(x0 - nmax);
            const float p1 = __expf(x1 - nmax);
            s0v[r] = p0; s1v[r] = p1;
            float rs = p0 + p1;
            rs += __shfl_xor(rs, 1);
            rs += __shfl_xor(rs, 2);
            rs += __shfl_xor(rs, 4);
            rs += __shfl_xor(rs, 8);
            runsum[r] = runsum[r] * scale + rs;
            o0[r] *= scale; o1[r] *= scale; o2[r] *= scale; o3[r] *= scale;
        }

        // ---- P: C-layout -> A-layout via per-wave LDS round-trip ----
        unsigned short* pw = pbuf[wave];
        #pragma unroll
        for (int r = 0; r < 8; ++r) {
            const int m = r + hf * 8;
            pw[m * 32 + row]      = f2bf(s0v[r]);
            pw[m * 32 + 16 + row] = f2bf(s1v[r]);
        }
        Frag16 pa;                                // same-wave DS ops are in-order
        {
            const unsigned short* pr = pw + row * 32;
            const int c0 = hf * 8;
            pa.q[0] = *(const uint4*)(pr + c0);
            pa.q[1] = *(const uint4*)(pr + c0 + 16);
        }

        // ---- O += P V  (4 n-tiles of 16, K-dim = 32 = full kv tile) ----
        #pragma unroll
        for (int nt = 0; nt < 4; ++nt) {
            Frag16 bvf;
            const unsigned short* vr = vtbuf + (nt * 16 + row) * 32 + hf * 8;
            bvf.q[0] = *(const uint4*)(vr);
            bvf.q[1] = *(const uint4*)(vr + 16);
            v8f& oo = (nt == 0 ? o0 : nt == 1 ? o1 : nt == 2 ? o2 : o3);
            oo = __builtin_amdgcn_wmma_f32_16x16x32_bf16(
                false, pa.v, false, bvf.v, (short)0, oo, false, false);
        }
        __syncthreads();
    }

    // ---- epilogue: normalize, write bf16 [seq][768] for the out-proj ----
    #pragma unroll
    for (int r = 0; r < 8; ++r) {
        const float inv = 1.0f / runsum[r];
        const int m = qbase + r + hf * 8;
        unsigned short* ob = Obuf + m * D_MODEL + h * D_HEAD;
        ob[row]      = f2bf(o0[r] * inv);
        ob[16 + row] = f2bf(o1[r] * inv);
        ob[32 + row] = f2bf(o2[r] * inv);
        ob[48 + row] = f2bf(o3[r] * inv);
    }
}

// -------------------------------------------------------------------------
// Kernel 3: output projection.  out = O @ Wo^T + bo  (bf16 A/B, f32 out)
// 32x32 register tile per wave, same as kernel 1.
// -------------------------------------------------------------------------
__global__ __launch_bounds__(256) void out_proj_kernel(
    const unsigned short* __restrict__ Obuf,
    const unsigned short* __restrict__ Wob,
    const float* __restrict__ bo,
    float* __restrict__ out)
{
    const int lane = threadIdx.x & 31;
    const int wave = threadIdx.x >> 5;
    const int row  = lane & 15;
    const int hf   = lane >> 4;

    const int mb = blockIdx.x * 32;
    const int nb = (blockIdx.y * 8 + wave) * 32;

    const unsigned short* arow0 = Obuf + (mb + row) * D_MODEL;
    const unsigned short* arow1 = Obuf + (mb + 16 + row) * D_MODEL;
    const unsigned short* brow0 = Wob + (nb + row) * D_MODEL;
    const unsigned short* brow1 = Wob + (nb + 16 + row) * D_MODEL;

    const int c0 = hf * 8;
    const int kb = hf * 8;

    v8f acc00 = {}, acc01 = {}, acc10 = {}, acc11 = {};
    for (int k0 = 0; k0 < D_MODEL; k0 += 32) {
        Frag16 a0, a1, b0, b1;
        a0.q[0] = *(const uint4*)(arow0 + k0 + c0);
        a0.q[1] = *(const uint4*)(arow0 + k0 + c0 + 16);
        a1.q[0] = *(const uint4*)(arow1 + k0 + c0);
        a1.q[1] = *(const uint4*)(arow1 + k0 + c0 + 16);
        b0.q[0] = *(const uint4*)(brow0 + k0 + kb);
        b0.q[1] = *(const uint4*)(brow0 + k0 + kb + 16);
        b1.q[0] = *(const uint4*)(brow1 + k0 + kb);
        b1.q[1] = *(const uint4*)(brow1 + k0 + kb + 16);
        acc00 = __builtin_amdgcn_wmma_f32_16x16x32_bf16(
            false, a0.v, false, b0.v, (short)0, acc00, false, false);
        acc01 = __builtin_amdgcn_wmma_f32_16x16x32_bf16(
            false, a0.v, false, b1.v, (short)0, acc01, false, false);
        acc10 = __builtin_amdgcn_wmma_f32_16x16x32_bf16(
            false, a1.v, false, b0.v, (short)0, acc10, false, false);
        acc11 = __builtin_amdgcn_wmma_f32_16x16x32_bf16(
            false, a1.v, false, b1.v, (short)0, acc11, false, false);
    }

    const int n0 = nb + row;
    const int n1 = nb + 16 + row;
    const float bb0 = bo[n0];
    const float bb1 = bo[n1];
    #pragma unroll
    for (int r = 0; r < 8; ++r) {
        const int m0 = mb + r + hf * 8;
        const int m1 = m0 + 16;
        out[m0 * D_MODEL + n0] = acc00[r] + bb0;
        out[m0 * D_MODEL + n1] = acc01[r] + bb1;
        out[m1 * D_MODEL + n0] = acc10[r] + bb0;
        out[m1 * D_MODEL + n1] = acc11[r] + bb1;
    }
}

// -------------------------------------------------------------------------
extern "C" void kernel_launch(void* const* d_in, const int* in_sizes, int n_in,
                              void* d_out, int out_size, void* d_ws, size_t ws_size,
                              hipStream_t stream) {
    const float* q  = (const float*)d_in[0];
    const float* k  = (const float*)d_in[1];
    const float* v  = (const float*)d_in[2];
    const float* wq = (const float*)d_in[3];
    const float* bq = (const float*)d_in[4];
    const float* wk = (const float*)d_in[5];
    const float* bk = (const float*)d_in[6];
    const float* wv = (const float*)d_in[7];
    const float* bv = (const float*)d_in[8];
    const float* wo = (const float*)d_in[9];
    const float* bo = (const float*)d_in[10];
    float* out = (float*)d_out;

    const size_t HEADBUF = (size_t)12 * SEQ * D_HEAD * sizeof(unsigned short); // 6 MB
    const size_t XBUF    = (size_t)SEQ * D_MODEL * sizeof(unsigned short);     // 6 MB
    const size_t WBUF    = (size_t)D_MODEL * D_MODEL * sizeof(unsigned short); // 1.125 MB

    char* ws = (char*)d_ws;
    unsigned short* Qh  = (unsigned short*)(ws);
    unsigned short* Kh  = (unsigned short*)(ws + HEADBUF);
    unsigned short* Vh  = (unsigned short*)(ws + 2 * HEADBUF);
    unsigned short* Ob  = (unsigned short*)(ws + 3 * HEADBUF);
    char* p = ws + 4 * HEADBUF;
    unsigned short* Xq  = (unsigned short*)(p);             p += XBUF;
    unsigned short* Xk  = (unsigned short*)(p);             p += XBUF;
    unsigned short* Xv  = (unsigned short*)(p);             p += XBUF;
    unsigned short* Wqb = (unsigned short*)(p);             p += WBUF;
    unsigned short* Wkb = (unsigned short*)(p);             p += WBUF;
    unsigned short* Wvb = (unsigned short*)(p);             p += WBUF;
    unsigned short* Wob = (unsigned short*)(p);             p += WBUF;

    const int nX8 = SEQ * D_MODEL / 8;        // 393216
    const int nW8 = D_MODEL * D_MODEL / 8;    // 73728
    cvt_bf16_kernel<<<(nX8 + 255) / 256, 256, 0, stream>>>(q,  Xq,  nX8);
    cvt_bf16_kernel<<<(nX8 + 255) / 256, 256, 0, stream>>>(k,  Xk,  nX8);
    cvt_bf16_kernel<<<(nX8 + 255) / 256, 256, 0, stream>>>(v,  Xv,  nX8);
    cvt_bf16_kernel<<<(nW8 + 255) / 256, 256, 0, stream>>>(wq, Wqb, nW8);
    cvt_bf16_kernel<<<(nW8 + 255) / 256, 256, 0, stream>>>(wk, Wkb, nW8);
    cvt_bf16_kernel<<<(nW8 + 255) / 256, 256, 0, stream>>>(wv, Wvb, nW8);
    cvt_bf16_kernel<<<(nW8 + 255) / 256, 256, 0, stream>>>(wo, Wob, nW8);

    // M-tiles: 4096/32 = 128; N: (blockIdx.y*8 + wave)*32 covers 768 with y=3.
    qkv_proj_kernel<<<dim3(128, 3, 3), 256, 0, stream>>>(
        Xq, Xk, Xv, Wqb, Wkb, Wvb, bq, bk, bv, Qh, Kh, Vh);
    attn_kernel<<<dim3(64, 12), 128, 0, stream>>>(Qh, Kh, Vh, Ob);
    out_proj_kernel<<<dim3(128, 3), 256, 0, stream>>>(Ob, Wob, bo, out);
}
